// SNNLanguageModel_79869211836656
// MI455X (gfx1250) — matrix operand order repaired and checked by reading, not compiled, not run on hardware
//
#include <hip/hip_runtime.h>

typedef __bf16 bf16;
typedef __attribute__((ext_vector_type(16))) __bf16 bf16x16;
typedef __attribute__((ext_vector_type(8)))  __bf16 bf16x8;
typedef __attribute__((ext_vector_type(8)))  float  v8f;
typedef __attribute__((ext_vector_type(4)))  int    v4i;

#define D_MODEL 1024
#define B_SZ    2
#define S_LEN   256
#define KF      32
#define T_LEN   (S_LEN*KF)        // 8192
#define NCH     (B_SZ*D_MODEL)    // 2048 channels (b,d)
#define NCHUNK  64
#define TCHUNK  (T_LEN/NCHUNK)    // 128
#define NLAYERS 4
#define VSIZE   6144
#define TBD     ((size_t)T_LEN*B_SZ*D_MODEL)   // 16,777,216
#define DD      ((size_t)D_MODEL*D_MODEL)

#define GEMM_BM   128
#define GEMM_BN   256
#define LDA_PITCH 40              // 80B row pitch -> conflict-free b128 frag reads

// Async global->LDS staging path (CDNA5 ASYNCcnt engine). The builtin's real
// signature (from the compiler diagnostic) takes int4 pointers: AS1 source,
// AS3 LDS destination.
#if defined(__AMDGCN__) && \
    __has_builtin(__builtin_amdgcn_global_load_async_to_lds_b128) && \
    __has_builtin(__builtin_amdgcn_s_wait_asynccnt)
#define USE_ASYNC_LDS 1
typedef __attribute__((address_space(1))) v4i as1_v4i;
typedef __attribute__((address_space(3))) v4i as3_v4i;
#else
#define USE_ASYNC_LDS 0
#if defined(__AMDGCN__)
#warning "gfx1250 async-to-LDS builtins not found; using register-staged fallback"
#endif
#endif

// ---------------------------------------------------------------------------
// Pack B (K x N, or transposed source) into WMMA-fragment-major bf16 layout:
// frag(kb, nt) is 32 lanes x 16 contiguous bf16; lane = (n&15) | ((kin&8)<<1),
// j = (kin&7) | ((kin&16)>>1)  -> matches V_WMMA 16-bit B 32x16 VGPR layout.
// ---------------------------------------------------------------------------
__global__ __launch_bounds__(256)
void pack_b_kernel(const float* __restrict__ W, bf16* __restrict__ Bp,
                   int K, int N, int transpose, int ld)
{
    int idx = blockIdx.x * blockDim.x + threadIdx.x;
    if (idx >= K * N) return;
    int k = idx / N, n = idx % N;
    float w = transpose ? W[(size_t)n * ld + k] : W[(size_t)k * ld + n];
    int kin  = k & 31;
    int lane = (n & 15) | ((kin & 8) << 1);
    int j    = (kin & 7) | ((kin & 16) >> 1);
    size_t dst = (((size_t)(k >> 5) * (N >> 4) + (n >> 4)) * 32 + lane) * 16 + j;
    Bp[dst] = (bf16)w;
}

// ---------------------------------------------------------------------------
// Encode: x[t][b][d] = embed[token_ids[b][t/K]][d]  (fp32 + bf16 mirror)
// ---------------------------------------------------------------------------
__global__ __launch_bounds__(256)
void encode_kernel(const int* __restrict__ tok, const float* __restrict__ embed,
                   float* __restrict__ xf, bf16* __restrict__ xh)
{
    int idx = blockIdx.x * blockDim.x + threadIdx.x;   // exactly T*B*D threads
    int d = idx & (D_MODEL - 1);
    int b = (idx >> 10) & (B_SZ - 1);
    int t = idx >> 11;
    int s = t >> 5;
    float v = embed[(size_t)tok[b * S_LEN + s] * D_MODEL + d];
    xf[idx] = v;
    xh[idx] = (bf16)v;
}

// ---------------------------------------------------------------------------
// WMMA GEMM: Cout[M,N] = A(bf16,MxK) * Bp(packed bf16 KxN) [+Cin] [+bias]
// block = 256 thr = 8 waves (2 M x 4 N); wave tile 64x64 = 4x4 WMMA tiles
// (16 v_wmma per k-step). A tile (128x32) staged in LDS; on CDNA5 the copy
// is done with GLOBAL_LOAD_ASYNC_TO_LDS_B128 + s_wait_asynccnt, LDS double
// buffered so the async engine overlaps the WMMA stream. B fragments stream
// from packed global (L2-resident weights) with global_prefetch_b8.
// ---------------------------------------------------------------------------
__global__ __launch_bounds__(256)
void gemm_bf16_wmma(const bf16* __restrict__ A, const bf16* __restrict__ Bp,
                    const float* __restrict__ Cin, const float* __restrict__ bias,
                    float* __restrict__ Cout, bf16* __restrict__ Obf,
                    int M, int N, int K)
{
#if USE_ASYNC_LDS
    __shared__ bf16 sA[2 * GEMM_BM * LDA_PITCH];
#else
    __shared__ bf16 sA[GEMM_BM * LDA_PITCH];
#endif
    const int tid  = threadIdx.x;
    const int lane = tid & 31;
    const int wid  = tid >> 5;
    const int wm   = wid & 1;              // 0..1  (M)
    const int wn   = wid >> 1;             // 0..3  (N)
    const int l15  = lane & 15;
    const int koff = (lane >> 4) << 3;     // 0 or 8 (A K-half select)
    const int blockM = blockIdx.x * GEMM_BM;
    const int blockN = blockIdx.y * GEMM_BN;
    const int NT16 = N >> 4;
    const int nk   = K >> 5;

    v8f acc[4][4];
#pragma unroll
    for (int mt = 0; mt < 4; ++mt)
#pragma unroll
        for (int ntl = 0; ntl < 4; ++ntl) acc[mt][ntl] = {};

#if USE_ASYNC_LDS
    // Each wave issues 2 async b128 ops per tile: 32 lanes x 16B x 2 = 1KB,
    // 8 waves cover the full 128x32 bf16 (8KB) tile.
    const int q0 = wid * 64 + lane;        // 16B-chunk id, j=0
    auto issue_tile = [&](int kb, int buf) {
#pragma unroll
        for (int j = 0; j < 2; ++j) {
            int q   = q0 + j * 32;         // 0..511
            int row = q >> 2;              // 0..127
            int cc  = (q & 3) * 8;         // bf16 col offset: 0,8,16,24
            const bf16* g = A + (size_t)(blockM + row) * K + (size_t)kb * 32 + cc;
            bf16* l = &sA[buf * (GEMM_BM * LDA_PITCH) + row * LDA_PITCH + cc];
            __builtin_amdgcn_global_load_async_to_lds_b128(
                (as1_v4i*)g, (as3_v4i*)l, 0, 0);
        }
    };
    issue_tile(0, 0);
#else
    // Fallback: register-staged double buffering
    const int ldRow = tid >> 1;            // 0..127
    const int ldCol = (tid & 1) * 16;      // 0 or 16
    const bf16* aG = A + (size_t)(blockM + ldRow) * K + ldCol;
    bf16x8 st0 = *(const bf16x8*)(aG + 0);
    bf16x8 st1 = *(const bf16x8*)(aG + 8);
#endif

    for (int kb = 0; kb < nk; ++kb) {
#if USE_ASYNC_LDS
        const int cur = kb & 1;
        __syncthreads();                   // everyone done reading buf cur^1
        if (kb + 1 < nk) {
            issue_tile(kb + 1, cur ^ 1);   // async engine fills next buffer
            __builtin_prefetch(
                Bp + (((size_t)(kb + 1) * NT16 + (blockN >> 4) + wn * 4) * 32 + lane) * 16,
                0, 1);
            __builtin_amdgcn_s_wait_asynccnt(2);  // tile kb landed (in-order)
        } else {
            __builtin_amdgcn_s_wait_asynccnt(0);
        }
        __syncthreads();                   // tile kb visible to all waves
        const bf16* sAc = &sA[cur * (GEMM_BM * LDA_PITCH)];
#else
        __syncthreads();
        *(bf16x8*)&sA[ldRow * LDA_PITCH + ldCol + 0] = st0;
        *(bf16x8*)&sA[ldRow * LDA_PITCH + ldCol + 8] = st1;
        __syncthreads();
        if (kb + 1 < nk) {
            const bf16* aN = aG + (size_t)(kb + 1) * 32;
            st0 = *(const bf16x8*)(aN + 0);
            st1 = *(const bf16x8*)(aN + 8);
            __builtin_prefetch(
                Bp + (((size_t)(kb + 1) * NT16 + (blockN >> 4) + wn * 4) * 32 + lane) * 16,
                0, 1);
        }
        const bf16* sAc = sA;
#endif

        // A fragments from LDS (16-bit A 16x32 layout: lane<16 K=0..7,16..23)
        bf16x16 af[4];
#pragma unroll
        for (int mt = 0; mt < 4; ++mt) {
            const bf16* ap = &sAc[(wm * 64 + mt * 16 + l15) * LDA_PITCH];
            union { bf16x16 v; bf16x8 h[2]; } u;
            u.h[0] = *(const bf16x8*)(ap + koff);
            u.h[1] = *(const bf16x8*)(ap + koff + 16);
            af[mt] = u.v;
        }
        // B fragments straight from packed global (32B contiguous per lane)
        bf16x16 bfr[4];
#pragma unroll
        for (int ntl = 0; ntl < 4; ++ntl) {
            int nt = (blockN >> 4) + wn * 4 + ntl;
            bfr[ntl] = *(const bf16x16*)(Bp + (((size_t)kb * NT16 + nt) * 32 + lane) * 16);
        }
#pragma unroll
        for (int mt = 0; mt < 4; ++mt)
#pragma unroll
            for (int ntl = 0; ntl < 4; ++ntl)
                acc[mt][ntl] = __builtin_amdgcn_wmma_f32_16x16x32_bf16(
                    false, af[mt], false, bfr[ntl], (short)0, acc[mt][ntl], false, false);
    }

    // Epilogue per documented 16x16 f32 C/D layout: row = r + (lane>=16)*8
    const int rowSel = (lane >> 4) << 3;
#pragma unroll
    for (int mt = 0; mt < 4; ++mt) {
#pragma unroll
        for (int ntl = 0; ntl < 4; ++ntl) {
            int col = blockN + wn * 64 + ntl * 16 + l15;
#pragma unroll
            for (int r = 0; r < 8; ++r) {
                int row = blockM + wm * 64 + mt * 16 + rowSel + r;
                size_t o = (size_t)row * N + col;
                float v = acc[mt][ntl][r];
                if (Cin)  v += Cin[o];
                if (bias) v += bias[col];
                Cout[o] = v;
                if (Obf) Obf[o] = (bf16)v;
            }
        }
    }
}

// ---------------------------------------------------------------------------
// PLIF scan, phase A: per-(chunk,channel) local scan from v=0, save chunk end
// ---------------------------------------------------------------------------
__global__ __launch_bounds__(256)
void scan_partial(const float* __restrict__ u, const float* __restrict__ beta_d,
                  float* __restrict__ vend)
{
    int idx   = blockIdx.x * blockDim.x + threadIdx.x;  // NCHUNK*NCH
    int c     = idx & (NCH - 1);
    int chunk = idx >> 11;
    int d     = c & (D_MODEL - 1);
    float beta = beta_d[d];
    const float* up = u + (size_t)chunk * TCHUNK * NCH + c;
    float v = 0.f;
#pragma unroll 4
    for (int t = 0; t < TCHUNK; ++t) v = beta * v + up[(size_t)t * NCH];
    vend[idx] = v;
}

// Phase B: carry across chunks; chunk decay = beta^128 via 7 squarings
__global__ __launch_bounds__(256)
void scan_carry(const float* __restrict__ beta_d, const float* __restrict__ vend,
                float* __restrict__ vstart)
{
    int c = blockIdx.x * blockDim.x + threadIdx.x;      // NCH
    int d = c & (D_MODEL - 1);
    float bp = beta_d[d];
#pragma unroll
    for (int i = 0; i < 7; ++i) bp *= bp;               // beta^(2^7) = beta^TCHUNK
    float v = 0.f;
    for (int k = 0; k < NCHUNK; ++k) {
        vstart[k * NCH + c] = v;
        v = bp * v + vend[k * NCH + c];
    }
}

// Phase C: recompute local scan from injected start, emit spike*vth
__global__ __launch_bounds__(256)
void scan_spike(const float* __restrict__ u, const float* __restrict__ beta_d,
                const float* __restrict__ vth_d, const float* __restrict__ vstart,
                bf16* __restrict__ spike_bf, float* __restrict__ vout_f)
{
    int idx   = blockIdx.x * blockDim.x + threadIdx.x;
    int c     = idx & (NCH - 1);
    int chunk = idx >> 11;
    int d     = c & (D_MODEL - 1);
    float beta = beta_d[d], vth = vth_d[d];
    bf16 vthb = (bf16)vth;
    float v = vstart[idx];
    const float* up = u + (size_t)chunk * TCHUNK * NCH + c;
    for (int t = 0; t < TCHUNK; ++t) {
        v = beta * v + up[(size_t)t * NCH];
        bool s = (v >= vth);
        size_t pos = (size_t)chunk * TCHUNK * NCH + (size_t)t * NCH + c;
        if (spike_bf) spike_bf[pos] = s ? vthb : (bf16)0.f;
        if (vout_f)   vout_f[pos]   = s ? vth  : 0.f;   // in-place over u is safe
    }
}

// ---------------------------------------------------------------------------
// RMSNorm over D=1024, one row per block; optional scalar alpha, f32/bf16 out
// ---------------------------------------------------------------------------
__global__ __launch_bounds__(256)
void rmsnorm_kernel(const float* __restrict__ X, const float* __restrict__ w,
                    const float* __restrict__ alpha, float* __restrict__ outF,
                    bf16* __restrict__ outB)
{
    __shared__ float red[8];
    int row = blockIdx.x;
    int tid = threadIdx.x;
    const float* xr = X + (size_t)row * D_MODEL;
    float vals[4];
    float ss = 0.f;
#pragma unroll
    for (int j = 0; j < 4; ++j) { float v = xr[tid + j * 256]; vals[j] = v; ss += v * v; }
#pragma unroll
    for (int off = 16; off >= 1; off >>= 1) ss += __shfl_xor(ss, off, 32);
    if ((tid & 31) == 0) red[tid >> 5] = ss;
    __syncthreads();
    if (tid < 32) {
        float r = (tid < 8) ? red[tid] : 0.f;
#pragma unroll
        for (int off = 4; off >= 1; off >>= 1) r += __shfl_xor(r, off, 32);
        if (tid == 0) red[0] = r;
    }
    __syncthreads();
    float rn = rsqrtf(red[0] * (1.f / D_MODEL) + 1e-6f);
    float a = alpha ? alpha[0] : 1.f;
#pragma unroll
    for (int j = 0; j < 4; ++j) {
        int dcol = tid + j * 256;
        float o = a * vals[j] * rn * w[dcol];
        if (outF) outF[(size_t)row * D_MODEL + dcol] = o;
        if (outB) outB[(size_t)row * D_MODEL + dcol] = (bf16)o;
    }
}

// K-frame mean: (T,b,D) -> (b*S+s, D) bf16
__global__ __launch_bounds__(256)
void aggregate_kernel(const float* __restrict__ vout, bf16* __restrict__ dec)
{
    int idx = blockIdx.x * blockDim.x + threadIdx.x;    // 512*1024
    int d = idx & (D_MODEL - 1);
    int r = idx >> 10;            // b*S + s
    int b = r >> 8;
    int s = r & (S_LEN - 1);
    float sum = 0.f;
#pragma unroll
    for (int k = 0; k < KF; ++k)
        sum += vout[((size_t)(s * KF + k) * B_SZ + b) * D_MODEL + d];
    dec[idx] = (bf16)(sum * (1.f / KF));
}

// ---------------------------------------------------------------------------
extern "C" void kernel_launch(void* const* d_in, const int* in_sizes, int n_in,
                              void* d_out, int out_size, void* d_ws, size_t ws_size,
                              hipStream_t stream)
{
    const int*   tok        = (const int*)  d_in[0];
    const float* embed      = (const float*)d_in[1];
    const float* W_in       = (const float*)d_in[2];
    const float* W_out      = (const float*)d_in[3];
    const float* layer_beta = (const float*)d_in[4];
    const float* layer_vth  = (const float*)d_in[5];
    const float* out_norm_w = (const float*)d_in[6];
    const float* out_beta   = (const float*)d_in[7];
    const float* out_vth    = (const float*)d_in[8];
    const float* out_alpha  = (const float*)d_in[9];
    const float* decode_w   = (const float*)d_in[10];
    const float* decode_b   = (const float*)d_in[11];
    const float* li_norm_w  = (const float*)d_in[12];
    float* out = (float*)d_out;

    // ---- workspace carve (256B aligned) ----
    char* p = (char*)d_ws;
    size_t used = 0;
    auto alloc = [&](size_t bytes) -> char* {
        char* r = p;
        size_t a = (bytes + 255) & ~(size_t)255;
        p += a; used += a;
        return r;
    };
    float* xf    = (float*)alloc(TBD * 4);             // residual stream fp32
    float* uu    = (float*)alloc(TBD * 4);             // u / h / v_out fp32
    bf16*  xh    = (bf16*) alloc(TBD * 2);             // x bf16 (GEMM A)
    bf16*  sh    = (bf16*) alloc(TBD * 2);             // spike*vth bf16
    bf16*  WinP  = (bf16*) alloc(NLAYERS * DD * 2);
    bf16*  WoutP = (bf16*) alloc(NLAYERS * DD * 2);
    bf16*  dwP   = (bf16*) alloc(DD * 2);
    bf16*  embP  = (bf16*) alloc((size_t)VSIZE * D_MODEL * 2);
    float* vend  = (float*)alloc((size_t)NCHUNK * NCH * 4);
    float* vstart= (float*)alloc((size_t)NCHUNK * NCH * 4);
    bf16*  decB  = (bf16*) alloc((size_t)B_SZ * S_LEN * D_MODEL * 2);
    float* h2    = (float*)alloc((size_t)B_SZ * S_LEN * D_MODEL * 4);
    bf16*  h2B   = (bf16*) alloc((size_t)B_SZ * S_LEN * D_MODEL * 2);
    if (used > ws_size) return;   // insufficient workspace: bail safely

    const int rows = T_LEN * B_SZ;                      // 16384
    const int nDD  = (int)DD;

    // ---- one-time weight packing ----
    for (int i = 0; i < NLAYERS; ++i) {
        pack_b_kernel<<<(nDD + 255) / 256, 256, 0, stream>>>(
            W_in + (size_t)i * DD, WinP + (size_t)i * DD, D_MODEL, D_MODEL, 0, D_MODEL);
        pack_b_kernel<<<(nDD + 255) / 256, 256, 0, stream>>>(
            W_out + (size_t)i * DD, WoutP + (size_t)i * DD, D_MODEL, D_MODEL, 0, D_MODEL);
    }
    pack_b_kernel<<<(nDD + 255) / 256, 256, 0, stream>>>(
        decode_w, dwP, D_MODEL, D_MODEL, 1, D_MODEL);           // decode_w^T
    pack_b_kernel<<<((int)(DD * VSIZE / D_MODEL) + 255) / 256, 256, 0, stream>>>(
        embed, embP, D_MODEL, VSIZE, 1, D_MODEL);               // embed^T

    // ---- encode ----
    encode_kernel<<<(int)(TBD / 256), 256, 0, stream>>>(tok, embed, xf, xh);

    // ---- SNN layers ----
    for (int i = 0; i < NLAYERS; ++i) {
        const float* bet = layer_beta + (size_t)i * D_MODEL;
        const float* vth = layer_vth  + (size_t)i * D_MODEL;
        gemm_bf16_wmma<<<dim3(rows / GEMM_BM, D_MODEL / GEMM_BN), 256, 0, stream>>>(
            xh, WinP + (size_t)i * DD, nullptr, nullptr, uu, nullptr,
            rows, D_MODEL, D_MODEL);
        scan_partial<<<NCHUNK * NCH / 256, 256, 0, stream>>>(uu, bet, vend);
        scan_carry<<<NCH / 256, 256, 0, stream>>>(bet, vend, vstart);
        scan_spike<<<NCHUNK * NCH / 256, 256, 0, stream>>>(uu, bet, vth, vstart, sh, nullptr);
        gemm_bf16_wmma<<<dim3(rows / GEMM_BM, D_MODEL / GEMM_BN), 256, 0, stream>>>(
            sh, WoutP + (size_t)i * DD, xf, nullptr, xf, xh,
            rows, D_MODEL, D_MODEL);
    }

    // ---- output neuron: rmsnorm -> scaled -> PLIF -> v_out (in place) ----
    rmsnorm_kernel<<<rows, 256, 0, stream>>>(xf, out_norm_w, out_alpha, uu, nullptr);
    scan_partial<<<NCHUNK * NCH / 256, 256, 0, stream>>>(uu, out_beta, vend);
    scan_carry<<<NCH / 256, 256, 0, stream>>>(out_beta, vend, vstart);
    scan_spike<<<NCHUNK * NCH / 256, 256, 0, stream>>>(uu, out_beta, out_vth, vstart,
                                                       nullptr, uu);

    // ---- K-frame mean -> decode proj -> norm -> LM head ----
    aggregate_kernel<<<(B_SZ * S_LEN * D_MODEL) / 256, 256, 0, stream>>>(uu, decB);
    gemm_bf16_wmma<<<dim3((B_SZ * S_LEN) / GEMM_BM, D_MODEL / GEMM_BN), 256, 0, stream>>>(
        decB, dwP, nullptr, decode_b, h2, nullptr, B_SZ * S_LEN, D_MODEL, D_MODEL);
    rmsnorm_kernel<<<B_SZ * S_LEN, 256, 0, stream>>>(h2, li_norm_w, nullptr, nullptr, h2B);
    gemm_bf16_wmma<<<dim3((B_SZ * S_LEN) / GEMM_BM, VSIZE / GEMM_BN), 256, 0, stream>>>(
        h2B, embP, nullptr, nullptr, out, nullptr, B_SZ * S_LEN, VSIZE, D_MODEL);
}
